// Block_39067022524586
// MI455X (gfx1250) — matrix operand order repaired and checked by reading, not compiled, not run on hardware
//
#include <hip/hip_runtime.h>
#include <hip/hip_bf16.h>

typedef __bf16 bf16_t;
typedef __attribute__((ext_vector_type(16))) __bf16 v16bf;
typedef __attribute__((ext_vector_type(8)))  float  v8f;
typedef __attribute__((ext_vector_type(4))) unsigned int u32x4;
typedef __attribute__((ext_vector_type(8))) int i32x8;
typedef __attribute__((ext_vector_type(4))) int i32x4;

constexpr int kB = 4, kS = 4096, kD = 1024, kH = 16, kE = 8, kK = 64;
constexpr int kHD = 64, kInter = 512;
constexpr int kN = kB * kS;                 // 16384 tokens
constexpr int kRowsPerExpert = kN / kE;     // 2048
constexpr float kEPS = 1.1920929e-07f;

// ---------------------------------------------------------------------------
// helpers
// ---------------------------------------------------------------------------
__device__ inline v16bf frag16(const bf16_t* p0, const bf16_t* p1) {
  union { uint4 u[2]; v16bf v; } t;
  t.u[0] = *(const uint4*)p0;
  t.u[1] = *(const uint4*)p1;
  return t.v;
}

// Issue a TDM 2D tile load: rows x 32 bf16 elements, row stride = strideElems.
// D# layout per CDNA5 ISA 8.3/8.4. Group0: count=1, lds_addr, global_addr,
// type=2. Group1: data_size=2B, tensor_dim0=strideElems, tensor_dim1=rows,
// tile_dim0=32, tile_dim1=rows, tensor_dim0_stride=strideElems.
// This toolchain exposes the 6-arg builtin:
//   (uint32x4 g0, int32x8 g1, int32x4 g2, int32x4 g3, int32x8 g4, i32 cpol)
__device__ inline void tdm_load_tile(const bf16_t* lds_dst, const bf16_t* gsrc,
                                     unsigned rows, unsigned strideElems) {
  const unsigned lds_off = (unsigned)(uintptr_t)lds_dst;   // addr[31:0] == LDS offset
  const unsigned long long ga = (unsigned long long)(uintptr_t)gsrc;
  u32x4 g0 = { 1u,                                   // count=1, is_restore=0
               lds_off,                               // lds_addr (bytes)
               (unsigned)ga,                          // global_addr[31:0]
               (unsigned)(ga >> 32) | (2u << 30) };   // global_addr[56:32] | type=2
  i32x8 g1 = {
    (int)(1u << 16),                                          // data_size = 1 (2 bytes)
    (int)((strideElems & 0xFFFFu) << 16),                     // tensor_dim0[15:0]
    (int)((strideElems >> 16) | ((rows & 0xFFFFu) << 16)),    // dim0 hi | tensor_dim1 lo
    (int)((rows >> 16) | (32u << 16)),                        // dim1 hi | tile_dim0 = 32
    (int)rows,                                                // tile_dim1 = rows, tile_dim2=0
    (int)strideElems,                                         // tensor_dim0_stride[31:0]
    0,                                                        // stride0 hi | stride1 lo
    0 };                                                      // stride1 hi
  i32x4 gz4 = { 0, 0, 0, 0 };
  i32x8 gz8 = { 0, 0, 0, 0, 0, 0, 0, 0 };
  __builtin_amdgcn_tensor_load_to_lds(g0, g1, gz4, gz4, gz8, 0);
}

// ---------------------------------------------------------------------------
// prep: f32 -> bf16 cast
// ---------------------------------------------------------------------------
__global__ void cast_bf16_kernel(const float* __restrict__ in, bf16_t* __restrict__ out, long n) {
  long i = (long)blockIdx.x * blockDim.x + threadIdx.x;
  if (i < n) out[i] = (bf16_t)in[i];
}

// out[a][b][c] = in[a][c][b], cast f32 -> bf16
__global__ void transpose_cast_kernel(const float* __restrict__ in, bf16_t* __restrict__ out,
                                      int A, int Bd, int Cd) {
  long i = (long)blockIdx.x * blockDim.x + threadIdx.x;
  long total = (long)A * Bd * Cd;
  if (i >= total) return;
  int c = (int)(i % Cd);
  long t = i / Cd;
  int b = (int)(t % Bd);
  int a = (int)(t / Bd);
  out[i] = (bf16_t)in[((long)a * Cd + c) * Bd + b];
}

// ---------------------------------------------------------------------------
// fused residual-mix + RMSNorm (one block per token, 256 thr x 4 elems)
// ---------------------------------------------------------------------------
__global__ void __launch_bounds__(256)
norm_kernel(const float* __restrict__ xin, const float* __restrict__ x0,
            const float* __restrict__ mix, float* __restrict__ x1out,
            bf16_t* __restrict__ nout) {
  __shared__ float red[256];
  const long p = blockIdx.x;
  const int tid = threadIdx.x;
  const float* xr = xin + p * kD;
  float vals[4];
  float ss = 0.f;
  const int base = tid * 4;
#pragma unroll
  for (int i = 0; i < 4; ++i) {
    int d = base + i;
    float v;
    if (mix) {
      v = mix[d] * xr[d] + mix[kD + d] * x0[p * kD + d];
    } else {
      v = xr[d];
    }
    vals[i] = v;
    ss += v * v;
  }
  red[tid] = ss;
  __syncthreads();
  for (int s = 128; s > 0; s >>= 1) {
    if (tid < s) red[tid] += red[tid + s];
    __syncthreads();
  }
  const float rn = rsqrtf(red[0] * (1.0f / kD) + kEPS);
#pragma unroll
  for (int i = 0; i < 4; ++i) {
    int d = base + i;
    if (x1out) x1out[p * kD + d] = vals[i];
    nout[p * kD + d] = (bf16_t)(vals[i] * rn);
  }
}

// ---------------------------------------------------------------------------
// gate = sigmoid(-softplus(beta[h]) * ||n_head - mu[h]||)   -> [N, H] f32
// ---------------------------------------------------------------------------
__global__ void gate_kernel(const bf16_t* __restrict__ nbf, const float* __restrict__ mu,
                            const float* __restrict__ beta, float* __restrict__ gate) {
  int idx = blockIdx.x * blockDim.x + threadIdx.x;   // 0..N*H-1
  if (idx >= kN * kH) return;
  int p = idx / kH;
  int h = idx % kH;
  const bf16_t* xr = nbf + (long)p * kD + h * kHD;
  const float* mur = mu + h * kHD;
  float ss = 0.f;
#pragma unroll 8
  for (int i = 0; i < kHD; ++i) {
    float d = (float)xr[i] - mur[i];
    ss += d * d;
  }
  float mag = sqrtf(ss);
  float sp = log1pf(expf(beta[h]));
  gate[idx] = 1.f / (1.f + expf(sp * mag));
}

// ---------------------------------------------------------------------------
// gather sorted rows: msorted[c] = mbf[sort_idx[c]]
// ---------------------------------------------------------------------------
__global__ void __launch_bounds__(256)
gather_kernel(const bf16_t* __restrict__ mbf, const int* __restrict__ sidx,
              bf16_t* __restrict__ msorted) {
  long c = blockIdx.x;
  long src = sidx[c];
  const uint2* s = (const uint2*)(mbf + src * kD);
  uint2* d = (uint2*)(msorted + c * kD);
  d[threadIdx.x] = s[threadIdx.x];
}

// ---------------------------------------------------------------------------
// Generic WMMA GEMM:  C[M,Nc] = A[M,K] @ Bw[Nc,K]^T   (bf16 in, f32 acc)
// Block: 256 thr = 8 waves, tile 128(M) x 64(N), K-step 32.
// A/B tiles staged into double-buffered LDS by the Tensor Data Mover
// (wave 0 issues tensor_load_to_lds for step i+1 while computing step i;
//  sync = s_wait_tensorcnt 0 + workgroup barrier).
// EPI: 0 = sigmoid -> outB    (q projection)
//      1 = outF = res + scale[j]*c          (attn residual, dense rows)
//      2 = relu(c)^2 -> outB   (expert fc, per-expert offsets)
//      3 = p = sidx[row]; outF[p] = res[p] + scale[j]*c  (expert proj scatter)
// ---------------------------------------------------------------------------
template <int EPI>
__global__ void __launch_bounds__(256)
gemm_wmma(const bf16_t* __restrict__ A, const bf16_t* __restrict__ Bw,
          int M, int Nc, int K,
          long strideAe, long strideBe, int rowsPerExpert,
          const float* __restrict__ res, const float* __restrict__ scale,
          float* __restrict__ outF, bf16_t* __restrict__ outB,
          const int* __restrict__ sidx, int ldOut) {
  __shared__ bf16_t sA[2][128 * 32];
  __shared__ bf16_t sB[2][64 * 32];

  const int e = blockIdx.z;
  A += (long)e * strideAe;
  Bw += (long)e * strideBe;
  if constexpr (EPI == 2) outB += (long)e * rowsPerExpert * ldOut;

  const int mBlk = blockIdx.x * 128;
  const int nBlk = blockIdx.y * 64;
  const int tid = threadIdx.x;
  const int lane = tid & 31;
  const int w = tid >> 5;
  const int m0 = (w >> 1) * 32;   // wave M origin in tile
  const int j0 = (w & 1) * 32;    // wave N origin in tile

  // fragment lane maps (ISA 7.12.2, bf16)
  const int rA = lane & 15;
  const int khA = (lane < 16) ? 0 : 8;
  const int cB = lane & 15;
  const int kbB = (lane < 16) ? 0 : 8;

  v8f acc[2][2] = {};

  const bf16_t* gA = A + (long)mBlk * K;
  const bf16_t* gB = Bw + (long)nBlk * K;
  const int nIter = K / 32;

  if (w == 0) {   // prologue: fill buffer 0 via TDM
    tdm_load_tile(&sA[0][0], gA, 128u, (unsigned)K);
    tdm_load_tile(&sB[0][0], gB, 64u, (unsigned)K);
  }

  for (int it = 0; it < nIter; ++it) {
    const int cur = it & 1;
    if (w == 0) __builtin_amdgcn_s_wait_tensorcnt(0);
    __syncthreads();                       // buffer `cur` ready for all waves
    if (w == 0 && it + 1 < nIter) {        // async-prefetch next k-step
      const int k0n = (it + 1) * 32;
      tdm_load_tile(&sA[cur ^ 1][0], gA + k0n, 128u, (unsigned)K);
      tdm_load_tile(&sB[cur ^ 1][0], gB + k0n, 64u, (unsigned)K);
    }

    v16bf af[2], bfv[2];
#pragma unroll
    for (int i = 0; i < 2; ++i) {
      const bf16_t* pa = &sA[cur][(m0 + i * 16 + rA) * 32 + khA];
      af[i] = frag16(pa, pa + 16);
      const bf16_t* pb = &sB[cur][(j0 + i * 16 + cB) * 32 + kbB];
      bfv[i] = frag16(pb, pb + 16);
    }
#pragma unroll
    for (int i = 0; i < 2; ++i)
#pragma unroll
      for (int j = 0; j < 2; ++j)
        acc[i][j] = __builtin_amdgcn_wmma_f32_16x16x32_bf16(
            false, af[i], false, bfv[j], (short)0, acc[i][j], false, false);
  }

  // epilogue (C layout: lane<16 -> M=r, N=lane ; lane>=16 -> M=r+8, N=lane-16)
  const int rowHalf = (lane < 16) ? 0 : 8;
  const int col = lane & 15;
#pragma unroll
  for (int i = 0; i < 2; ++i)
#pragma unroll
    for (int j = 0; j < 2; ++j) {
      const int gmB = mBlk + m0 + i * 16 + rowHalf;
      const int gj = nBlk + j0 + j * 16 + col;
#pragma unroll
      for (int r = 0; r < 8; ++r) {
        const int gm = gmB + r;
        const float v = acc[i][j][r];
        if constexpr (EPI == 0) {
          outB[(long)gm * ldOut + gj] = (bf16_t)(1.f / (1.f + expf(-v)));
        } else if constexpr (EPI == 1) {
          const long idx = (long)gm * ldOut + gj;
          outF[idx] = res[idx] + scale[gj] * v;
        } else if constexpr (EPI == 2) {
          const float rr = v > 0.f ? v : 0.f;
          outB[(long)gm * ldOut + gj] = (bf16_t)(rr * rr);
        } else {
          const int p = sidx[e * rowsPerExpert + gm];
          const long idx = (long)p * ldOut + gj;
          outF[idx] = res[idx] + scale[gj] * v;
        }
      }
    }
}

// ---------------------------------------------------------------------------
// Causal grouped conv as implicit-im2col WMMA GEMM.
// K-order: k = t*64 + i (taps outer, in-channels inner).
//   A[m,k] = n[b, s0+m-63+t, g*64+i]  -> contiguous 8-elem per-lane reads
//   B[k,o] = wT[o, t, i]              -> contiguous 8-elem per-lane reads
// Epilogue fuses y = q * gate * h  (bf16 out).
// Grid: (S/32, H, B); block 256 thr = 8 waves = 2(M) x 4(out-ch) 16x16 tiles.
// ---------------------------------------------------------------------------
__global__ void __launch_bounds__(256)
conv_wmma(const bf16_t* __restrict__ nbf, const bf16_t* __restrict__ wT,
          const bf16_t* __restrict__ qbf, const float* __restrict__ gate,
          bf16_t* __restrict__ ybf) {
  const int b = blockIdx.z;
  const int g = blockIdx.y;
  const int sTile = blockIdx.x * 32;
  const int tid = threadIdx.x;
  const int lane = tid & 31;
  const int w = tid >> 5;
  const int m0 = sTile + (w >> 2) * 16;   // token tile origin
  const int j0 = (w & 3) * 16;            // out-channel tile within group

  const int rA = lane & 15;
  const int khA = (lane < 16) ? 0 : 8;
  const int colB = lane & 15;
  const int kbB = (lane < 16) ? 0 : 8;

  const bf16_t* xbase = nbf + ((long)b * kS) * kD + g * 64;
  const bf16_t* wbase = wT + (long)(g * 64 + j0 + colB) * (kK * kHD);

  v8f acc = {};
  const uint4 z = make_uint4(0u, 0u, 0u, 0u);

  for (int t = 0; t < kK; ++t) {
    const int srow = m0 + rA - (kK - 1) + t;
    const bool ok = srow >= 0;
    const bf16_t* xr = xbase + (long)srow * kD;
#pragma unroll
    for (int i0 = 0; i0 < kHD; i0 += 32) {
      union { uint4 u[2]; v16bf v; } a;
      a.u[0] = ok ? *(const uint4*)(xr + i0 + khA) : z;
      a.u[1] = ok ? *(const uint4*)(xr + i0 + khA + 16) : z;
      const bf16_t* wp = wbase + t * kHD + i0 + kbB;
      union { uint4 u[2]; v16bf v; } bb;
      bb.u[0] = *(const uint4*)wp;
      bb.u[1] = *(const uint4*)(wp + 16);
      acc = __builtin_amdgcn_wmma_f32_16x16x32_bf16(
          false, a.v, false, bb.v, (short)0, acc, false, false);
    }
  }

  const int rowHalf = (lane < 16) ? 0 : 8;
  const int col = lane & 15;
#pragma unroll
  for (int r = 0; r < 8; ++r) {
    const int s = m0 + rowHalf + r;
    const long tok = (long)b * kS + s;
    const int ch = g * 64 + j0 + col;
    const float qv = (float)qbf[tok * kD + ch];
    const float gv = gate[tok * kH + g];
    ybf[tok * kD + ch] = (bf16_t)(qv * gv * acc[r]);
  }
}

// ---------------------------------------------------------------------------
// host launch
// ---------------------------------------------------------------------------
extern "C" void kernel_launch(void* const* d_in, const int* in_sizes, int n_in,
                              void* d_out, int out_size, void* d_ws, size_t ws_size,
                              hipStream_t stream) {
  const float* x        = (const float*)d_in[0];
  const float* x0       = (const float*)d_in[1];
  const float* mu       = (const float*)d_in[2];
  const float* beta     = (const float*)d_in[3];
  const float* q_proj_w = (const float*)d_in[4];
  const float* conv_w   = (const float*)d_in[5];
  const float* out_proj = (const float*)d_in[6];
  const float* fc_w     = (const float*)d_in[7];
  const float* proj_w   = (const float*)d_in[8];
  const float* attn_sc  = (const float*)d_in[9];
  const float* mlp_sc   = (const float*)d_in[10];
  const float* residmix = (const float*)d_in[11];
  const int*   sort_idx = (const int*)d_in[12];
  float* out = (float*)d_out;

  char* wsb = (char*)d_ws;
  size_t off = 0;
  auto take = [&](size_t bytes) { char* p = wsb + off; off += (bytes + 255) & ~(size_t)255; return p; };

  float*  x1      = (float*)take((size_t)kN * kD * 4);
  bf16_t* nbf     = (bf16_t*)take((size_t)kN * kD * 2);      // reused as mbf
  bf16_t* qbf     = (bf16_t*)take((size_t)kN * kD * 2);      // reused as msorted
  bf16_t* ybf     = (bf16_t*)take((size_t)kN * kD * 2);      // reused as hdn
  float*  gateb   = (float*)take((size_t)kN * kH * 4);
  bf16_t* wq_bf   = (bf16_t*)take((size_t)kD * kD * 2);
  bf16_t* wo_bf   = (bf16_t*)take((size_t)kD * kD * 2);
  bf16_t* wconvT  = (bf16_t*)take((size_t)kD * kK * kHD * 2);
  bf16_t* fcT     = (bf16_t*)take((size_t)kE * kInter * kD * 2);
  bf16_t* projT   = (bf16_t*)take((size_t)kE * kD * kInter * 2);
  bf16_t* mbf     = nbf;
  bf16_t* msorted = qbf;
  bf16_t* hdn     = ybf;
  (void)ws_size; (void)n_in; (void)in_sizes; (void)out_size;

  // ---- weight prep ----
  {
    long n = (long)kD * kD;
    cast_bf16_kernel<<<(n + 255) / 256, 256, 0, stream>>>(q_proj_w, wq_bf, n);
    cast_bf16_kernel<<<(n + 255) / 256, 256, 0, stream>>>(out_proj, wo_bf, n);
    long nc = (long)kD * kK * kHD;   // conv: (O,I,T) -> (O,T,I)
    transpose_cast_kernel<<<(nc + 255) / 256, 256, 0, stream>>>(conv_w, wconvT, kD, kK, kHD);
    long nf = (long)kE * kInter * kD; // fc: (E,D,I) -> (E,I,D)
    transpose_cast_kernel<<<(nf + 255) / 256, 256, 0, stream>>>(fc_w, fcT, kE, kInter, kD);
    long np = (long)kE * kD * kInter; // proj: (E,I,D) -> (E,D,I)
    transpose_cast_kernel<<<(np + 255) / 256, 256, 0, stream>>>(proj_w, projT, kE, kD, kInter);
  }

  // ---- x1 = mix0*x + mix1*x0 ; n = rmsnorm(x1) ----
  norm_kernel<<<kN, 256, 0, stream>>>(x, x0, residmix, x1, nbf);

  // ---- gate[N,H] ----
  gate_kernel<<<(kN * kH) / 256, 256, 0, stream>>>(nbf, mu, beta, gateb);

  // ---- q = sigmoid(n @ Wq^T) ----
  gemm_wmma<0><<<dim3(kN / 128, kD / 64, 1), 256, 0, stream>>>(
      nbf, wq_bf, kN, kD, kD, 0, 0, 0,
      nullptr, nullptr, nullptr, qbf, nullptr, kD);

  // ---- y = q * gate * conv(n) ----
  conv_wmma<<<dim3(kS / 32, kH, kB), 256, 0, stream>>>(nbf, wconvT, qbf, gateb, ybf);

  // ---- x2 = x1 + attn_scale * (y @ Wo^T)  -> stored in d_out ----
  gemm_wmma<1><<<dim3(kN / 128, kD / 64, 1), 256, 0, stream>>>(
      ybf, wo_bf, kN, kD, kD, 0, 0, 0,
      x1, attn_sc, out, nullptr, nullptr, kD);

  // ---- m = rmsnorm(x2) ----
  norm_kernel<<<kN, 256, 0, stream>>>(out, nullptr, nullptr, nullptr, mbf);

  // ---- gather by sort_idx ----
  gather_kernel<<<kN, 256, 0, stream>>>(mbf, sort_idx, msorted);

  // ---- hdn = relu(msorted @ fc)^2  (per expert) ----
  gemm_wmma<2><<<dim3(kRowsPerExpert / 128, kInter / 64, kE), 256, 0, stream>>>(
      msorted, fcT, kRowsPerExpert, kInter, kD,
      (long)kRowsPerExpert * kD, (long)kInter * kD, kRowsPerExpert,
      nullptr, nullptr, nullptr, hdn, nullptr, kInter);

  // ---- out[sort_idx[c]] = x2 + mlp_scale * (hdn @ proj)  (per expert) ----
  gemm_wmma<3><<<dim3(kRowsPerExpert / 128, kD / 64, kE), 256, 0, stream>>>(
      hdn, projT, kRowsPerExpert, kD, kInter,
      (long)kRowsPerExpert * kInter, (long)kD * kInter, kRowsPerExpert,
      out, mlp_sc, out, nullptr, sort_idx, kD);
}
